// MixerBlock_12524124635797
// MI455X (gfx1250) — matrix operand order repaired and checked by reading, not compiled, not run on hardware
//
#include <hip/hip_runtime.h>
#include <cstdint>

typedef __attribute__((ext_vector_type(16))) __bf16 v16bf;
typedef __attribute__((ext_vector_type(8)))  __bf16 v8bf;
typedef __attribute__((ext_vector_type(8)))  float  v8f;
typedef __attribute__((ext_vector_type(4)))  float  f32x4;

// ---------------------------------------------------------------------------
// CDNA5 async copy: global -> LDS, 16B per lane, tracked by ASYNCcnt.
// ---------------------------------------------------------------------------
static __device__ __forceinline__ void async_ld16(uint32_t lds_byte_addr, const void* gptr) {
    asm volatile("global_load_async_to_lds_b128 %0, %1, off"
                 :: "v"(lds_byte_addr), "v"(gptr)
                 : "memory");
}
static __device__ __forceinline__ void wait_async0() {
    asm volatile("s_wait_asynccnt 0" ::: "memory");
}

// ---------------------------------------------------------------------------
// Fragment loaders (CDNA5 wave32 WMMA layouts, ISA 7.12.2)
// A 16x32 bf16: lane<16 -> M=lane, K in {0..7, 16..23}; lane>=16 -> K in {8..15, 24..31}
// B 32x16 bf16: lane<16 -> N=lane, K in {0..7, 16..23}; lane>=16 -> K in {8..15, 24..31}
// ---------------------------------------------------------------------------
static __device__ __forceinline__ v16bf load_fragA(const __bf16* row, int hs) {
    const int k0 = hs << 3;
    v8bf c0 = *(const v8bf*)(row + k0);
    v8bf c1 = *(const v8bf*)(row + k0 + 16);
    v16bf r;
#pragma unroll
    for (int i = 0; i < 8; ++i) { r[i] = c0[i]; r[i + 8] = c1[i]; }
    return r;
}

static __device__ __forceinline__ v16bf load_fragB(const __bf16* row, int hs) {
    const int k0 = hs << 3;
    v8bf c0 = *(const v8bf*)(row + k0);
    v8bf c1 = *(const v8bf*)(row + k0 + 16);
    v16bf r;
#pragma unroll
    for (int i = 0; i < 8; ++i) { r[i] = c0[i]; r[i + 8] = c1[i]; }
    return r;
}

// ---------------------------------------------------------------------------
// Generic bf16x3 GEMM: D = A(M,K) * B(N,K)^T with fused epilogues.
//   mode 0: YAT epilogue    -> writes bf16 hi/lo split of yat(dot) to Ohi/Olo (row stride Nreal)
//   mode 1: token linear    -> y1[b,p,c] = x + dot + b2[p]; writes y1 fp32 + bf16 hi/lo (transposed store)
//   mode 2: channel linear  -> out[b,p,c] = y1 + dot + b4[c]
// Block: 256 threads (8 waves), tile 128(M) x 128(N), wave tile 64x32, K step 32.
// LDS: double-buffered async-DMA staging of {Ahi, Alo, Bhi, Blo} tiles.
// ---------------------------------------------------------------------------

// Per-buffer LDS layout (bytes): 4 arrays of 128 rows x 40 bf16 (80B row stride)
#define TILE_BYTES   10240u                  // 128*40*2
#define BUF_BYTES    40960u                  // 4 tiles
#define OFF_AL       10240u
#define OFF_BH       20480u
#define OFF_BL       30720u

__global__ __launch_bounds__(256) void gemm_bf16x3(
    const __bf16* __restrict__ Ahi, const __bf16* __restrict__ Alo,
    const __bf16* __restrict__ Bhi, const __bf16* __restrict__ Blo,
    int K,
    const float* __restrict__ anorm, const float* __restrict__ wnorm,
    const float* __restrict__ bias,  const float* __restrict__ alphaPtr,
    float outF, int mode, int Nreal,
    __bf16* __restrict__ Ohi, __bf16* __restrict__ Olo,
    float* __restrict__ Of, const float* __restrict__ resid)
{
    extern __shared__ __align__(16) char smem_raw[];   // 2 * BUF_BYTES

    const int tid  = threadIdx.x;
    const int lane = tid & 31;
    const int wave = tid >> 5;
    const int wm   = (wave & 1) * 64;
    const int wn   = (wave >> 1) * 32;
    const size_t bm = (size_t)blockIdx.y * 128;
    const size_t bn = (size_t)blockIdx.x * 128;
    const int fr = lane & 15;
    const int hs = lane >> 4;

    // Per-thread staging chunk coordinates (2 x 16B chunks per tile per thread)
    const int q0 = tid, q1 = tid + 256;
    const int r0 = q0 >> 2, c0s = (q0 & 3) << 3;
    const int r1 = q1 >> 2, c1s = (q1 & 3) << 3;
    const uint32_t lds_base = (uint32_t)(uintptr_t)smem_raw;
    const uint32_t l0 = lds_base + (uint32_t)(r0 * 40 + c0s) * 2u;
    const uint32_t l1 = lds_base + (uint32_t)(r1 * 40 + c1s) * 2u;

    v8f acc[4][2];
#pragma unroll
    for (int mi = 0; mi < 4; ++mi)
#pragma unroll
        for (int ni = 0; ni < 2; ++ni)
#pragma unroll
            for (int e = 0; e < 8; ++e) acc[mi][ni][e] = 0.0f;

    // Async-issue one K-tile into LDS buffer `buf`
    auto issue = [&](int buf, int k0) {
        const uint32_t bofs = (uint32_t)buf * BUF_BYTES;
        const size_t a0 = (bm + r0) * (size_t)K + k0 + c0s;
        const size_t a1 = (bm + r1) * (size_t)K + k0 + c1s;
        const size_t b0 = (bn + r0) * (size_t)K + k0 + c0s;
        const size_t b1 = (bn + r1) * (size_t)K + k0 + c1s;
        async_ld16(l0 + bofs,           Ahi + a0);
        async_ld16(l1 + bofs,           Ahi + a1);
        async_ld16(l0 + bofs + OFF_AL,  Alo + a0);
        async_ld16(l1 + bofs + OFF_AL,  Alo + a1);
        async_ld16(l0 + bofs + OFF_BH,  Bhi + b0);
        async_ld16(l1 + bofs + OFF_BH,  Bhi + b1);
        async_ld16(l0 + bofs + OFF_BL,  Blo + b0);
        async_ld16(l1 + bofs + OFF_BL,  Blo + b1);
    };

    const int nsteps = K >> 5;
    issue(0, 0);

    for (int it = 0; it < nsteps; ++it) {
        const int cur = it & 1;
        wait_async0();          // my fills of buf[cur] are in LDS
        __syncthreads();        // everyone's fills done; everyone done reading buf[cur^1]
        if (it + 1 < nsteps) issue(cur ^ 1, (it + 1) << 5);

        const __bf16* sb  = (const __bf16*)(smem_raw + cur * BUF_BYTES);
        const __bf16* sAh = sb;
        const __bf16* sAl = sb + 5120;
        const __bf16* sBh = sb + 10240;
        const __bf16* sBl = sb + 15360;

        v16bf aH[4], aL[4], bH[2], bL[2];
#pragma unroll
        for (int mi = 0; mi < 4; ++mi) {
            aH[mi] = load_fragA(&sAh[(wm + mi * 16 + fr) * 40], hs);
            aL[mi] = load_fragA(&sAl[(wm + mi * 16 + fr) * 40], hs);
        }
#pragma unroll
        for (int ni = 0; ni < 2; ++ni) {
            bH[ni] = load_fragB(&sBh[(wn + ni * 16 + fr) * 40], hs);
            bL[ni] = load_fragB(&sBl[(wn + ni * 16 + fr) * 40], hs);
        }
#pragma unroll
        for (int mi = 0; mi < 4; ++mi)
#pragma unroll
            for (int ni = 0; ni < 2; ++ni) {
                v8f a = acc[mi][ni];
                a = __builtin_amdgcn_wmma_f32_16x16x32_bf16(false, aH[mi], false, bH[ni], (short)0, a, false, false);
                a = __builtin_amdgcn_wmma_f32_16x16x32_bf16(false, aH[mi], false, bL[ni], (short)0, a, false, false);
                a = __builtin_amdgcn_wmma_f32_16x16x32_bf16(false, aL[mi], false, bH[ni], (short)0, a, false, false);
                acc[mi][ni] = a;
            }
    }

    // ---------------- epilogue ----------------
    float scale = 0.0f;
    if (mode == 0) {
        const float a  = alphaPtr[0];
        const float sb = outF / logf(outF + 1.0f);
        scale = powf(sqrtf(sb), a);
    }

#pragma unroll
    for (int mi = 0; mi < 4; ++mi) {
#pragma unroll
        for (int ni = 0; ni < 2; ++ni) {
            const int n  = (int)bn + wn + ni * 16 + fr;
            const int m0 = (int)bm + wm + mi * 16 + hs * 8;
            v8f d = acc[mi][ni];
            if (mode == 0) {
                const float wv = wnorm[n];
                const float bb = bias[n];
#pragma unroll
                for (int i = 0; i < 8; ++i) {
                    const float dd  = d[i];
                    const float dot = dd + bb;
                    const float dsq = wv + anorm[m0 + i] - 2.0f * dd;
                    const float v   = scale * dot * dot / (dsq + 0.1f);
                    const __bf16 h  = (__bf16)v;
                    const size_t idx = (size_t)(m0 + i) * (size_t)Nreal + n;
                    Ohi[idx] = h;
                    Olo[idx] = (__bf16)(v - (float)h);
                }
            } else if (mode == 1) {
                const int p = n;                      // output column = patch index
                if (p < 196) {
                    const float bb = bias[p];
                    const int b_ = m0 / 768;
                    const int c0 = m0 % 768;          // 8-aligned, no batch crossing
                    const size_t idx = ((size_t)b_ * 196 + p) * 768 + c0;
                    f32x4 rr0 = *(const f32x4*)&resid[idx];
                    f32x4 rr1 = *(const f32x4*)&resid[idx + 4];
                    f32x4 o0, o1;
                    v8bf h8, l8;
#pragma unroll
                    for (int i = 0; i < 8; ++i) {
                        const float rv = (i < 4) ? rr0[i] : rr1[i - 4];
                        const float v  = rv + d[i] + bb;
                        if (i < 4) o0[i] = v; else o1[i - 4] = v;
                        const __bf16 h = (__bf16)v;
                        h8[i] = h;
                        l8[i] = (__bf16)(v - (float)h);
                    }
                    *(f32x4*)&Of[idx]     = o0;
                    *(f32x4*)&Of[idx + 4] = o1;
                    *(v8bf*)&Ohi[idx] = h8;
                    *(v8bf*)&Olo[idx] = l8;
                }
            } else {
                const float bb = bias[n];
#pragma unroll
                for (int i = 0; i < 8; ++i) {
                    const size_t idx = (size_t)(m0 + i) * 768 + n;
                    Of[idx] = resid[idx] + d[i] + bb;
                }
            }
        }
    }
}

// ---------------------------------------------------------------------------
// x (B,196,768) -> xT (B,768,224) as bf16 hi/lo (K padded 196->224 with zeros)
// ---------------------------------------------------------------------------
__global__ __launch_bounds__(256) void token_transpose_split(
    const float* __restrict__ x, __bf16* __restrict__ xTh, __bf16* __restrict__ xTl)
{
    __shared__ float tile[32][33];
    const int b = blockIdx.z, pt = blockIdx.y, ct = blockIdx.x;
    const int p0 = pt * 32, c0 = ct * 32;
    const int tc = threadIdx.x & 31, tr = threadIdx.x >> 5;
#pragma unroll
    for (int j = 0; j < 4; ++j) {
        const int pr = tr + 8 * j;
        const int p  = p0 + pr;
        tile[pr][tc] = (p < 196) ? x[((size_t)b * 196 + p) * 768 + c0 + tc] : 0.0f;
    }
    __syncthreads();
#pragma unroll
    for (int j = 0; j < 4; ++j) {
        const int cr = tr + 8 * j;
        const float v = tile[tc][cr];
        const size_t idx = ((size_t)b * 768 + c0 + cr) * 224 + p0 + tc;
        const __bf16 h = (__bf16)v;
        xTh[idx] = h;
        xTl[idx] = (__bf16)(v - (float)h);
    }
}

// ||x||^2 over the patch dim for each (b,c): strided but fully coalesced reads.
__global__ __launch_bounds__(256) void colnorm_x(const float* __restrict__ x,
                                                 float* __restrict__ xnorm)
{
    const int b = blockIdx.y;
    const int c = blockIdx.x * 256 + threadIdx.x;
    const float* p = x + (size_t)b * 196 * 768 + c;
    float s = 0.0f;
    for (int q = 0; q < 196; ++q) { const float v = p[(size_t)q * 768]; s += v * v; }
    xnorm[(size_t)b * 768 + c] = s;
}

// Row norms of a contiguous (R,K) matrix: one wave per row.
__global__ __launch_bounds__(256) void rownorm(const float* __restrict__ in,
                                               float* __restrict__ out, int R, int K)
{
    const int row  = blockIdx.x * 8 + (threadIdx.x >> 5);
    const int lane = threadIdx.x & 31;
    if (row >= R) return;
    const float* p = in + (size_t)row * K;
    float s = 0.0f;
    for (int k = lane; k < K; k += 32) { const float v = p[k]; s += v * v; }
#pragma unroll
    for (int m = 16; m; m >>= 1) s += __shfl_xor(s, m, 32);
    if (lane == 0) out[row] = s;
}

// fp32 (N,K) -> bf16 hi/lo (rows padded to Nalloc, cols padded to Kp), zero fill.
__global__ __launch_bounds__(256) void split_pad(const float* __restrict__ in,
                                                 __bf16* __restrict__ hi,
                                                 __bf16* __restrict__ lo,
                                                 int N, int K, int Kp, long total)
{
    const long i = (long)blockIdx.x * 256 + threadIdx.x;
    if (i >= total) return;
    const int r = (int)(i / Kp);
    const int k = (int)(i % Kp);
    const float v = (r < N && k < K) ? in[(size_t)r * K + k] : 0.0f;
    const __bf16 h = (__bf16)v;
    hi[i] = h;
    lo[i] = (__bf16)(v - (float)h);
}

// ---------------------------------------------------------------------------
extern "C" void kernel_launch(void* const* d_in, const int* in_sizes, int n_in,
                              void* d_out, int out_size, void* d_ws, size_t ws_size,
                              hipStream_t stream)
{
    (void)in_sizes; (void)n_in; (void)out_size; (void)ws_size;
    const float* x  = (const float*)d_in[0];
    const float* tw = (const float*)d_in[1];
    const float* tb = (const float*)d_in[2];
    const float* ta = (const float*)d_in[3];
    const float* w2 = (const float*)d_in[4];
    const float* b2 = (const float*)d_in[5];
    const float* cw = (const float*)d_in[6];
    const float* cb = (const float*)d_in[7];
    const float* ca = (const float*)d_in[8];
    const float* w4 = (const float*)d_in[9];
    const float* b4 = (const float*)d_in[10];
    float* out = (float*)d_out;
    char* ws = (char*)d_ws;

    constexpr int Bn = 64, P = 196, C = 768, T = 384, CM = 3072;
    constexpr int MT = Bn * C;   // 49152 token rows (b,c)
    constexpr int KT = 224;      // 196 padded to mult of 32
    constexpr int MC = Bn * P;   // 12544 channel rows (b,p)

    size_t o = 0;
    auto alloc = [&](size_t bytes) { size_t r = (o + 255) & ~(size_t)255; o = r + bytes; return r; };

    const size_t o_xnorm = alloc((size_t)MT * 4);
    const size_t o_twn   = alloc((size_t)T * 4);
    const size_t o_cwn   = alloc((size_t)CM * 4);
    const size_t o_y1n   = alloc((size_t)MC * 4);
    const size_t o_twh   = alloc((size_t)T * KT * 2);
    const size_t o_twl   = alloc((size_t)T * KT * 2);
    const size_t o_w2h   = alloc((size_t)256 * T * 2);   // w2 rows padded 196->256
    const size_t o_w2l   = alloc((size_t)256 * T * 2);
    const size_t o_cwh   = alloc((size_t)CM * C * 2);
    const size_t o_cwl   = alloc((size_t)CM * C * 2);
    const size_t o_w4h   = alloc((size_t)C * CM * 2);
    const size_t o_w4l   = alloc((size_t)C * CM * 2);
    const size_t o_y1    = alloc((size_t)MC * C * 4);
    const size_t o_y1h   = alloc((size_t)MC * C * 2);
    const size_t o_y1l   = alloc((size_t)MC * C * 2);

    // Overlaid big region: {xT hi/lo, h_t hi/lo} (phase 1) vs {h2 hi/lo} (phase 2)
    const size_t sz_xT = (size_t)MT * KT * 2;
    const size_t sz_ht = (size_t)MT * T * 2;
    const size_t sz_h2 = (size_t)MC * CM * 2;
    const size_t phase1 = 2 * sz_xT + 2 * sz_ht;
    const size_t phase2 = 2 * sz_h2;
    const size_t o_big = alloc(phase1 > phase2 ? phase1 : phase2);
    const size_t o_xTh = o_big;
    const size_t o_xTl = o_big + sz_xT;
    const size_t o_hth = o_big + 2 * sz_xT;
    const size_t o_htl = o_big + 2 * sz_xT + sz_ht;
    const size_t o_h2h = o_big;
    const size_t o_h2l = o_big + sz_h2;

    float*  xnorm = (float*)(ws + o_xnorm);
    float*  twn = (float*)(ws + o_twn);
    float*  cwn = (float*)(ws + o_cwn);
    float*  y1n = (float*)(ws + o_y1n);
    __bf16* twh = (__bf16*)(ws + o_twh);
    __bf16* twl = (__bf16*)(ws + o_twl);
    __bf16* w2h = (__bf16*)(ws + o_w2h);
    __bf16* w2l = (__bf16*)(ws + o_w2l);
    __bf16* cwh = (__bf16*)(ws + o_cwh);
    __bf16* cwl = (__bf16*)(ws + o_cwl);
    __bf16* w4h = (__bf16*)(ws + o_w4h);
    __bf16* w4l = (__bf16*)(ws + o_w4l);
    float*  y1  = (float*)(ws + o_y1);
    __bf16* y1h = (__bf16*)(ws + o_y1h);
    __bf16* y1l = (__bf16*)(ws + o_y1l);
    __bf16* xTh = (__bf16*)(ws + o_xTh);
    __bf16* xTl = (__bf16*)(ws + o_xTl);
    __bf16* hth = (__bf16*)(ws + o_hth);
    __bf16* htl = (__bf16*)(ws + o_htl);
    __bf16* h2h = (__bf16*)(ws + o_h2h);
    __bf16* h2l = (__bf16*)(ws + o_h2l);

    const dim3 blk(256);
    const size_t gemm_lds = 2u * BUF_BYTES;   // 80 KB double-buffered staging
    auto cdiv = [](long a, long b) { return (unsigned)((a + b - 1) / b); };

    // Stage 1: transpose+split x, norms, weight splits
    token_transpose_split<<<dim3(C / 32, KT / 32, Bn), blk, 0, stream>>>(x, xTh, xTl);
    colnorm_x<<<dim3(C / 256, Bn), blk, 0, stream>>>(x, xnorm);
    rownorm<<<dim3(cdiv(T, 8)), blk, 0, stream>>>(tw, twn, T, P);
    rownorm<<<dim3(cdiv(CM, 8)), blk, 0, stream>>>(cw, cwn, CM, C);
    split_pad<<<dim3(cdiv((long)T * KT, 256)), blk, 0, stream>>>(tw, twh, twl, T, P, KT, (long)T * KT);
    split_pad<<<dim3(cdiv(256L * T, 256)), blk, 0, stream>>>(w2, w2h, w2l, P, T, T, 256L * T);
    split_pad<<<dim3(cdiv((long)CM * C, 256)), blk, 0, stream>>>(cw, cwh, cwl, CM, C, C, (long)CM * C);
    split_pad<<<dim3(cdiv((long)C * CM, 256)), blk, 0, stream>>>(w4, w4h, w4l, C, CM, CM, (long)C * CM);

    // Stage 2: token YAT GEMM (M=49152, N=384, K=224) -> h_t hi/lo
    gemm_bf16x3<<<dim3(T / 128, MT / 128), blk, gemm_lds, stream>>>(
        xTh, xTl, twh, twl, KT, xnorm, twn, tb, ta, (float)T, 0, T,
        hth, htl, nullptr, nullptr);

    // Stage 3: token linear GEMM (M=49152, Npad=256, K=384) -> y1 (+ fused split)
    gemm_bf16x3<<<dim3(256 / 128, MT / 128), blk, gemm_lds, stream>>>(
        hth, htl, w2h, w2l, T, nullptr, nullptr, b2, nullptr, 0.0f, 1, P,
        y1h, y1l, y1, x);

    // Stage 4: ||y1||^2 rows
    rownorm<<<dim3(cdiv(MC, 8)), blk, 0, stream>>>(y1, y1n, MC, C);

    // Stage 5: channel YAT GEMM (M=12544, N=3072, K=768) -> h2 hi/lo
    gemm_bf16x3<<<dim3(CM / 128, MC / 128), blk, gemm_lds, stream>>>(
        y1h, y1l, cwh, cwl, C, y1n, cwn, cb, ca, (float)CM, 0, CM,
        h2h, h2l, nullptr, nullptr);

    // Stage 6: channel linear GEMM (M=12544, N=768, K=3072) -> out
    gemm_bf16x3<<<dim3(C / 128, MC / 128), blk, gemm_lds, stream>>>(
        h2h, h2l, w4h, w4l, CM, nullptr, nullptr, b4, nullptr, 0.0f, 2, 0,
        nullptr, nullptr, out, y1);
}
